// PhysarumRouter_23055384445087
// MI455X (gfx1250) — compile-verified, matrix operand order checked by reading, not run on hardware
//
#include <hip/hip_runtime.h>
#include <hip/hip_bf16.h>
#include <math.h>

#define B_    64
#define P_    576
#define D_    768
#define N_    65      // GROUPS + 1
#define NPAD  80      // padded to 5 M-tiles of 16
#define G_    64
#define S_    8
#define GW_   24

typedef __attribute__((ext_vector_type(16))) __bf16 v16bf;
typedef __attribute__((ext_vector_type(8)))  float  v8f;

union BfPack { unsigned int u[8]; v16bf v; };

// ---- output layout (floats, flat, reference return order) ----
#define O_KEEP   0          // (64,577)
#define O_PATCH  36928      // (64,576)
#define O_GID    73792      // (64,576)
#define O_GS     110656     // (64,576)
#define O_DH     147520     // (64,9,65,65)
#define O_QH     2581120    // (64,8,65,65)
#define O_RT     4744320    // (64,64,64)
#define O_AUX    5006464    // 2 scalars

// ---- workspace layout (bytes) ----
#define WS_NODES 0u                      // 64*80*768 bf16
#define WS_W     7864320u                // 768*768 bf16
#define WS_PROJ  9043968u                // 64*80*768 f32
#define WS_PROJN 24772608u               // 64*80*768 bf16
#define WS_SP    32636928u               // 64 f32
#define WS_ST    32637184u               // 64 f32

__device__ __forceinline__ unsigned short f2bf(float f) {
  unsigned int x = __float_as_uint(f);
  unsigned int r = (x + 0x7FFFu + ((x >> 16) & 1u)) >> 16;  // RNE
  return (unsigned short)r;
}

// Load one 16x32 bf16 tile operand per CDNA5 WMMA layout.
// base: row-major [16 x 768] bf16; r = lane&15 (row), kh = (lane>>4)*8.
__device__ __forceinline__ void load_tile(const unsigned short* __restrict__ base,
                                          int r, int kh, int k0, BfPack& pk) {
  const unsigned short* p = base + (size_t)r * D_ + k0 + kh;
#pragma unroll
  for (int v = 0; v < 4; ++v) {
    pk.u[v]     = *(const unsigned int*)(p + 2 * v);        // K = kh + 2v, +1
    pk.u[v + 4] = *(const unsigned int*)(p + 16 + 2 * v);   // K = 16 + kh + 2v, +1
  }
}

// ---------------- kernel 1: build nodes (cls + 9-patch group means), bf16, pad rows zeroed
__global__ void k_nodes(const float* __restrict__ tokens, const float* __restrict__ cls,
                        unsigned short* __restrict__ nodes) {
  const int row = blockIdx.x;   // 0..79
  const int b   = blockIdx.y;
  unsigned short* dst = nodes + ((size_t)b * NPAD + row) * D_;
  if (row == 0) {
    const float* src = cls + (size_t)b * D_;
    for (int d = threadIdx.x; d < D_; d += blockDim.x) dst[d] = f2bf(src[d]);
  } else if (row < N_) {
    const int g = row - 1;
    const int r0 = (g >> 3) * 3, c0 = (g & 7) * 3;
    const float* tb = tokens + (size_t)b * P_ * D_;
    for (int d = threadIdx.x; d < D_; d += blockDim.x) {
      float s = 0.f;
#pragma unroll
      for (int rr = 0; rr < 3; ++rr)
#pragma unroll
        for (int cc = 0; cc < 3; ++cc) {
          int p = (r0 + rr) * GW_ + (c0 + cc);
          s += tb[(size_t)p * D_ + d];
        }
      dst[d] = f2bf(s * (1.f / 9.f));
    }
  } else {
    for (int d = threadIdx.x; d < D_; d += blockDim.x) dst[d] = 0;
  }
}

// ---------------- kernel 2: W f32 -> bf16
__global__ void k_wcvt(const float* __restrict__ W, unsigned short* __restrict__ Wb) {
  size_t i = (size_t)blockIdx.x * blockDim.x + threadIdx.x;
  if (i < (size_t)D_ * D_) Wb[i] = f2bf(W[i]);
}

// ---------------- kernel 3: proj = nodes @ W^T via v_wmma_f32_16x16x32_bf16
// one wave per block; each wave owns one 16-row M strip x 4 N-tiles (16x64 output)
__global__ void __launch_bounds__(32) k_gemm_proj(const unsigned short* __restrict__ nodes,
                                                  const unsigned short* __restrict__ Wb,
                                                  float* __restrict__ proj) {
  const int lane = threadIdx.x;
  const int b = blockIdx.z, mt = blockIdx.y, ntb = blockIdx.x * 4;
  const unsigned short* A = nodes + (size_t)b * NPAD * D_ + (size_t)(mt * 16) * D_;
  const int r  = lane & 15;
  const int kh = (lane >> 4) * 8;
  v8f acc[4] = {};
  for (int k0 = 0; k0 < D_; k0 += 32) {
    BfPack a;
    load_tile(A, r, kh, k0, a);
#pragma unroll
    for (int t = 0; t < 4; ++t) {
      BfPack bm;
      load_tile(Wb + (size_t)(ntb + t) * 16 * D_, r, kh, k0, bm);
      acc[t] = __builtin_amdgcn_wmma_f32_16x16x32_bf16(
          false, a.v, false, bm.v, (short)0, acc[t], false, false);
    }
  }
  const int hi = lane >> 4, n = lane & 15;
  float* outp = proj + (size_t)b * NPAD * D_;
#pragma unroll
  for (int t = 0; t < 4; ++t)
#pragma unroll
    for (int v = 0; v < 8; ++v) {
      int row = mt * 16 + v + 8 * hi;
      outp[(size_t)row * D_ + (ntb + t) * 16 + n] = acc[t][v];
    }
}

// ---------------- kernel 4: L2-normalize proj rows -> bf16
__global__ void __launch_bounds__(256) k_norm(const float* __restrict__ proj,
                                              unsigned short* __restrict__ projn) {
  const int row = blockIdx.x, b = blockIdx.y;
  const float* pr = proj + ((size_t)b * NPAD + row) * D_;
  unsigned short* dst = projn + ((size_t)b * NPAD + row) * D_;
  __shared__ float red[256];
  if (row >= N_) {
    for (int d = threadIdx.x; d < D_; d += blockDim.x) dst[d] = 0;
    return;
  }
  float ss = 0.f;
  for (int d = threadIdx.x; d < D_; d += blockDim.x) { float v = pr[d]; ss += v * v; }
  red[threadIdx.x] = ss; __syncthreads();
  for (int o = 128; o > 0; o >>= 1) {
    if (threadIdx.x < o) red[threadIdx.x] += red[threadIdx.x + o];
    __syncthreads();
  }
  float inv = 1.f / fmaxf(sqrtf(red[0]), 1e-12f);
  for (int d = threadIdx.x; d < D_; d += blockDim.x) dst[d] = f2bf(pr[d] * inv);
}

// ---------------- kernel 5: gram -> C0 = floor(exp(-d2)), written to d_hist[:,0]
__global__ void __launch_bounds__(32) k_gram(const unsigned short* __restrict__ projn,
                                             float* __restrict__ dhist) {
  const int lane = threadIdx.x;
  const int b = blockIdx.z, mt = blockIdx.y, nt = blockIdx.x;
  const unsigned short* base = projn + (size_t)b * NPAD * D_;
  const unsigned short* A  = base + (size_t)(mt * 16) * D_;
  const unsigned short* Bp = base + (size_t)(nt * 16) * D_;
  const int r = lane & 15, kh = (lane >> 4) * 8;
  v8f acc = {};
  for (int k0 = 0; k0 < D_; k0 += 32) {
    BfPack a, bm;
    load_tile(A, r, kh, k0, a);
    load_tile(Bp, r, kh, k0, bm);
    acc = __builtin_amdgcn_wmma_f32_16x16x32_bf16(
        false, a.v, false, bm.v, (short)0, acc, false, false);
  }
  const int hi = lane >> 4, n = lane & 15;
#pragma unroll
  for (int v = 0; v < 8; ++v) {
    int row = mt * 16 + v + 8 * hi, col = nt * 16 + n;
    if (row < N_ && col < N_) {
      float g = acc[v];
      float d2 = fmaxf(2.f - 2.f * g, 0.f);         // rows unit-norm
      float val = expf(-d2);                        // TAU = 1
      val = (row == col) ? 0.f : fmaxf(val, 1e-4f); // _with_floor
      dhist[(size_t)b * 9 * (N_ * N_) + (size_t)row * N_ + col] = val;
    }
  }
}

// ---------------- kernel 6: 8 physarum steps; GE solve in LDS; writes d_hist/q_hist/routing + aux partials
__global__ void __launch_bounds__(256) k_iter(float* __restrict__ dh_all,
                                              float* __restrict__ qh_all,
                                              float* __restrict__ rout_all,
                                              float* __restrict__ ws_sp,
                                              float* __restrict__ ws_st) {
  const int b = blockIdx.x, tid = threadIdx.x;
  __shared__ float Cm[N_ * 66];
  __shared__ float Ms[N_ * 67];
  __shared__ float fvec[N_], pvec[N_], degs[N_];
  __shared__ float red[256];
  float* dh = dh_all + (size_t)b * 9 * (N_ * N_);
  float* qh = qh_all + (size_t)b * 8 * (N_ * N_);

  for (int idx = tid; idx < N_ * N_; idx += 256)
    Cm[(idx / N_) * 66 + idx % N_] = dh[idx];
  __syncthreads();

  float stab = 0.f;
  for (int s = 0; s < S_; ++s) {
    // degrees
    for (int i = tid; i < N_; i += 256) {
      float sum = 0.f;
      for (int j = 0; j < N_; ++j) sum += Cm[i * 66 + j];
      degs[i] = sum;
    }
    __syncthreads();
    // augmented system: (deg+eps)I - C | source
    for (int idx = tid; idx < N_ * 66; idx += 256) {
      int i = idx / 66, j = idx % 66;
      float v;
      if (j == N_)      v = (i == 0) ? 1.f : (-1.f / 64.f);
      else              v = ((i == j) ? (degs[i] + 1e-4f) : 0.f) - Cm[i * 66 + j];
      Ms[i * 67 + j] = v;
    }
    __syncthreads();
    // forward elimination (diagonally dominant -> no pivoting)
    for (int k = 0; k < N_ - 1; ++k) {
      float piv = Ms[k * 67 + k];
      for (int i = k + 1 + tid; i < N_; i += 256) fvec[i] = Ms[i * 67 + k] / piv;
      __syncthreads();
      int nrows = N_ - 1 - k, ncols = N_ - k;   // j = k+1..65 (incl. rhs)
      int tot = nrows * ncols;
      for (int t = tid; t < tot; t += 256) {
        int i = k + 1 + t / ncols, j = k + 1 + t % ncols;
        Ms[i * 67 + j] -= fvec[i] * Ms[k * 67 + j];
      }
      __syncthreads();
    }
    // back substitution
    for (int k = N_ - 1; k >= 0; --k) {
      if (tid == 0) pvec[k] = Ms[k * 67 + 65] / Ms[k * 67 + k];
      __syncthreads();
      float pk = pvec[k];
      for (int i = tid; i < k; i += 256) Ms[i * 67 + 65] -= Ms[i * 67 + k] * pk;
      __syncthreads();
    }
    // flow + conductivity update (symmetry preserved exactly)
    for (int idx = tid; idx < N_ * N_; idx += 256) {
      int i = idx / N_, j = idx % N_;
      float c = Cm[i * 66 + j];
      float fl = c * (pvec[i] - pvec[j]);
      qh[(size_t)s * (N_ * N_) + idx] = fl;
      float af = fabsf(fl);
      float rr = af / (1.f + af);                 // MU = 1
      float cn = c + 0.1f * (rr - 0.1f * c);      // DT, GAMMA
      cn = fmaxf(cn, 0.f);
      cn = (i == j) ? 0.f : fmaxf(cn, 1e-4f);
      stab += fabsf(cn - c);
      Cm[i * 66 + j] = cn;
      dh[(size_t)(s + 1) * (N_ * N_) + idx] = cn;
    }
    __syncthreads();
  }
  // routing = final_d[1:,1:] row-normalized
  for (int g = tid; g < G_; g += 256) {
    float rs = 0.f;
    for (int h = 0; h < G_; ++h) rs += Cm[(1 + g) * 66 + 1 + h];
    degs[g] = fmaxf(rs, 1e-6f);
  }
  __syncthreads();
  float* rout = rout_all + (size_t)b * G_ * G_;
  float sp = 0.f;
  for (int idx = tid; idx < G_ * G_; idx += 256) {
    int g = idx / G_, h = idx % G_;
    float v = Cm[(1 + g) * 66 + 1 + h];
    sp += v;
    rout[idx] = v / degs[g];
  }
  red[tid] = sp; __syncthreads();
  for (int o = 128; o > 0; o >>= 1) { if (tid < o) red[tid] += red[tid + o]; __syncthreads(); }
  if (tid == 0) ws_sp[b] = red[0];
  __syncthreads();
  red[tid] = stab; __syncthreads();
  for (int o = 128; o > 0; o >>= 1) { if (tid < o) red[tid] += red[tid + o]; __syncthreads(); }
  if (tid == 0) ws_st[b] = red[0];
}

// ---------------- kernel 7: scores, top-k (rank counting), masks, gid, group_scores
__global__ void __launch_bounds__(256) k_scores(const float* __restrict__ local_scores,
                                                const float* __restrict__ qh_all,
                                                float* __restrict__ out_keep,
                                                float* __restrict__ out_patch,
                                                float* __restrict__ out_gid,
                                                float* __restrict__ out_gs) {
  const int b = blockIdx.x, tid = threadIdx.x;
  __shared__ float gf[G_], gsn[G_], lsv[P_], ts[P_], sc[4];
  const float* q = qh_all + ((size_t)b * 8 + 7) * (N_ * N_);
  for (int g = tid; g < G_; g += 256) {
    const float* row = q + (size_t)(1 + g) * N_;
    float s = 0.f;
    for (int j = 0; j < N_; ++j) s += fabsf(row[j]);
    gf[g] = s;
  }
  for (int p = tid; p < P_; p += 256) lsv[p] = local_scores[(size_t)b * P_ + p];
  __syncthreads();
  if (tid == 0) {
    float m = 0.f; for (int g = 0; g < G_; ++g) m += gf[g]; m /= (float)G_;
    float v = 0.f; for (int g = 0; g < G_; ++g) { float d = gf[g] - m; v += d * d; }
    v /= (float)(G_ - 1);
    sc[0] = m; sc[1] = fmaxf(sqrtf(v), 1e-6f);
    float lm = 0.f; for (int p = 0; p < P_; ++p) lm += lsv[p]; lm /= (float)P_;
    float lv = 0.f; for (int p = 0; p < P_; ++p) { float d = lsv[p] - lm; lv += d * d; }
    lv /= (float)(P_ - 1);
    sc[2] = lm; sc[3] = fmaxf(sqrtf(lv), 1e-6f);
  }
  __syncthreads();
  for (int g = tid; g < G_; g += 256) gsn[g] = (gf[g] - sc[0]) / sc[1];
  __syncthreads();
  for (int p = tid; p < P_; p += 256) {
    int row = p / GW_, col = p % GW_;
    int g = (row / 3) * 8 + (col / 3);
    float gs = gsn[g];
    ts[p] = 1.0f * gs + 0.5f * ((lsv[p] - sc[2]) / sc[3]);
    out_gs[(size_t)b * P_ + p]  = gs;
    out_gid[(size_t)b * P_ + p] = (float)g;
  }
  __syncthreads();
  for (int p = tid; p < P_; p += 256) {
    float s0 = ts[p];
    int rank = 0;
    for (int qi = 0; qi < P_; ++qi) {
      float s1 = ts[qi];
      rank += (s1 > s0) || (s1 == s0 && qi < p);
    }
    float keep = (rank < 288) ? 1.f : 0.f;        // KEEP_RATIO*P
    out_patch[(size_t)b * P_ + p] = keep;
    out_keep[(size_t)b * (P_ + 1) + 1 + p] = keep;
  }
  if (tid == 0) out_keep[(size_t)b * (P_ + 1)] = 1.f;
}

// ---------------- kernel 8: aux scalars
__global__ void k_aux(const float* __restrict__ ws_sp, const float* __restrict__ ws_st,
                      float* __restrict__ out) {
  __shared__ float a[64], c[64];
  int t = threadIdx.x;
  a[t] = ws_sp[t]; c[t] = ws_st[t];
  __syncthreads();
  for (int o = 32; o > 0; o >>= 1) {
    if (t < o) { a[t] += a[t + o]; c[t] += c[t + o]; }
    __syncthreads();
  }
  if (t == 0) { out[0] = a[0] / 64.f; out[1] = c[0] / 64.f; }
}

extern "C" void kernel_launch(void* const* d_in, const int* in_sizes, int n_in,
                              void* d_out, int out_size, void* d_ws, size_t ws_size,
                              hipStream_t stream) {
  const float* tokens = (const float*)d_in[0];
  const float* cls    = (const float*)d_in[1];
  const float* W      = (const float*)d_in[2];
  const float* ls     = (const float*)d_in[3];
  (void)in_sizes; (void)n_in; (void)out_size; (void)ws_size;

  char* ws = (char*)d_ws;
  unsigned short* nodes = (unsigned short*)(ws + WS_NODES);
  unsigned short* Wb    = (unsigned short*)(ws + WS_W);
  float*          proj  = (float*)(ws + WS_PROJ);
  unsigned short* projn = (unsigned short*)(ws + WS_PROJN);
  float*          wssp  = (float*)(ws + WS_SP);
  float*          wsst  = (float*)(ws + WS_ST);

  float* out = (float*)d_out;

  k_nodes<<<dim3(NPAD, B_), 256, 0, stream>>>(tokens, cls, nodes);
  k_wcvt <<<(D_ * D_ + 255) / 256, 256, 0, stream>>>(W, Wb);
  k_gemm_proj<<<dim3(12, 5, B_), 32, 0, stream>>>(nodes, Wb, proj);
  k_norm <<<dim3(NPAD, B_), 256, 0, stream>>>(proj, projn);
  k_gram <<<dim3(5, 5, B_), 32, 0, stream>>>(projn, out + O_DH);
  k_iter <<<B_, 256, 0, stream>>>(out + O_DH, out + O_QH, out + O_RT, wssp, wsst);
  k_scores<<<B_, 256, 0, stream>>>(ls, out + O_QH, out + O_KEEP, out + O_PATCH,
                                   out + O_GID, out + O_GS);
  k_aux<<<1, 64, 0, stream>>>(wssp, wsst, out + O_AUX);
}
